// VoxelQFormer_58849641890276
// MI455X (gfx1250) — compile-verified, matrix operand order checked by reading, not run on hardware
//
#include <hip/hip_runtime.h>

// ---------------------------------------------------------------------------
// VoxelQFormer decoder for MI455X (gfx1250, wave32, WMMA bf16 -> f32 acc).
// Fragment loads are two ds_load_b128 per operand (ISA 16-bit layout:
// lane-half lh holds contiguous K = lh*8..+7 and 16+lh*8..+7).
// ---------------------------------------------------------------------------

#define HIDDIM  768
#define NQCNT   80
#define BATCH   8
#define NVOX    2048
#define TTXT    64
#define NHEADS  12
#define HDIM    64
#define DFFDIM  3072
#define NLAYERS 4

typedef __bf16 bf16;
typedef __attribute__((ext_vector_type(16))) __bf16 v16bf;
typedef __attribute__((ext_vector_type(8)))  __bf16 v8bf;
typedef __attribute__((ext_vector_type(8)))  float  v8f;

__device__ inline v8f wmma_bf16(v16bf a, v16bf b, v8f c) {
  return __builtin_amdgcn_wmma_f32_16x16x32_bf16(false, a, false, b,
                                                 (short)0, c, false, false);
}

// Load one 16x32 operand row (A row or B^T row) as two 16B vector loads.
__device__ inline v16bf ld_frag(const bf16* p, int lh) {
  union { v16bf v; v8bf h[2]; } u;
  u.h[0] = *(const v8bf*)(p + lh * 8);        // K = lh*8 .. +7   -> vgpr 0..3
  u.h[1] = *(const v8bf*)(p + 16 + lh * 8);   // K = 16+lh*8 ..+7 -> vgpr 4..7
  return u.v;
}

__device__ inline v8bf zero8() {
  v8bf z;
  for (int j = 0; j < 8; ++j) z[j] = (bf16)0.f;
  return z;
}

// ---------------------------------------------------------------------------
// Small elementwise kernels
// ---------------------------------------------------------------------------
__global__ __launch_bounds__(256) void rel_kernel(
    const float* __restrict__ sv, const float* __restrict__ pos,
    const float* __restrict__ ori, float* __restrict__ rel, int total) {
  int i = blockIdx.x * 256 + threadIdx.x;
  if (i >= total) return;
  int b = i / NVOX;
  float rx = sv[(size_t)i * 6 + 0] - pos[b * 3 + 0];
  float ry = sv[(size_t)i * 6 + 1] - pos[b * 3 + 1];
  float rz = sv[(size_t)i * 6 + 2] - pos[b * 3 + 2];
  float ox = ori[b * 2 + 0], oy = ori[b * 2 + 1];
  float inv = rsqrtf(ox * ox + oy * oy);
  float x = ox * inv, y = oy * inv;
  rel[(size_t)i * 3 + 0] = rx * y - ry * x;
  rel[(size_t)i * 3 + 1] = rx * x + ry * y;
  rel[(size_t)i * 3 + 2] = rz;
}

__global__ __launch_bounds__(256) void voxmask_kernel(
    const int* __restrict__ vm, const int* __restrict__ m2,
    unsigned char* __restrict__ kp, int total) {
  int i = blockIdx.x * 256 + threadIdx.x;
  if (i >= total) return;
  int b = i / (2 * NVOX);
  int j = i % (2 * NVOX);
  int n = j & (NVOX - 1);
  int pad = (vm[b * NVOX + n] == 0);
  if (j >= NVOX) pad |= (m2[b] == 0);
  kp[i] = (unsigned char)pad;
}

__global__ __launch_bounds__(256) void txtmask_kernel(
    const int* __restrict__ tm, unsigned char* __restrict__ kp, int total) {
  int i = blockIdx.x * 256 + threadIdx.x;
  if (i >= total) return;
  kp[i] = (tm[i] == 0) ? (unsigned char)1 : (unsigned char)0;
}

__global__ __launch_bounds__(256) void addpos_kernel(
    const float* __restrict__ Q, const float* __restrict__ qpos,
    float* __restrict__ outp, int total) {
  int i = blockIdx.x * 256 + threadIdx.x;
  if (i >= total) return;
  int c = i % HIDDIM;
  int nq = (i / HIDDIM) % NQCNT;
  outp[i] = Q[i] + qpos[nq * HIDDIM + c];
}

// residual add + LayerNorm over 768, one block per row
__global__ __launch_bounds__(256) void add_ln_kernel(
    float* __restrict__ Q, const float* __restrict__ Tm,
    const float* __restrict__ sc, const float* __restrict__ bs) {
  const int row = blockIdx.x;
  const int tid = threadIdx.x;
  __shared__ float red[256];
  float x0 = Q[(size_t)row * 768 + tid + 0]   + Tm[(size_t)row * 768 + tid + 0];
  float x1 = Q[(size_t)row * 768 + tid + 256] + Tm[(size_t)row * 768 + tid + 256];
  float x2 = Q[(size_t)row * 768 + tid + 512] + Tm[(size_t)row * 768 + tid + 512];
  red[tid] = x0 + x1 + x2;
  __syncthreads();
  for (int st = 128; st > 0; st >>= 1) {
    if (tid < st) red[tid] += red[tid + st];
    __syncthreads();
  }
  float mean = red[0] * (1.0f / 768.0f);
  __syncthreads();
  float d0 = x0 - mean, d1 = x1 - mean, d2 = x2 - mean;
  red[tid] = d0 * d0 + d1 * d1 + d2 * d2;
  __syncthreads();
  for (int st = 128; st > 0; st >>= 1) {
    if (tid < st) red[tid] += red[tid + st];
    __syncthreads();
  }
  float inv = rsqrtf(red[0] * (1.0f / 768.0f) + 1e-5f);
  Q[(size_t)row * 768 + tid + 0]   = d0 * inv * sc[tid + 0]   + bs[tid + 0];
  Q[(size_t)row * 768 + tid + 256] = d1 * inv * sc[tid + 256] + bs[tid + 256];
  Q[(size_t)row * 768 + tid + 512] = d2 * inv * sc[tid + 512] + bs[tid + 512];
}

// ---------------------------------------------------------------------------
// bf16-WMMA GEMM:  C[M,N] = A[M,K] @ W[K,N] + bias
// Block: 256 threads (8 waves), tile 128(M) x 64(N), K-steps of 32.
// Wave grid 4x2; wave tile 32x32 = 4 WMMA accumulators (2 A-frags x 2 B-frags
// per k-step -> 2 ds_load_b128 per WMMA).
// A_BF16: A bf16 vs fp32.  OUT_MODE: 0=f32, 1=bf16, 2=relu f32.
// ROPE: fused epilogue (+tgt-embed, rotary, bf16 store into memory_vox).
// ---------------------------------------------------------------------------
template <int A_BF16, int OUT_MODE, int ROPE>
__global__ __launch_bounds__(256) void gemm_kernel(
    const void* __restrict__ A_, const float* __restrict__ W,
    const float* __restrict__ bias, void* __restrict__ C_,
    int M, int N, int K,
    const int* __restrict__ tgt_mask, const float* __restrict__ tgt_row,
    const float* __restrict__ rel, const float* __restrict__ inv_freq,
    int mem_off) {
  __shared__ __align__(16) bf16 As[128][40];  // [m][k], 80B rows (16B aligned)
  __shared__ __align__(16) bf16 Bs[64][40];   // [n][k] (W tile transposed)
  const int tid = threadIdx.x;
  const int lane = tid & 31;
  const int wave = tid >> 5;
  const int r  = lane & 15;
  const int lh = lane >> 4;
  const int wm = wave >> 1;  // 0..3 -> rows wm*32
  const int wn = wave & 1;   // 0..1 -> cols wn*32
  const int rowBase = blockIdx.y * 128;
  const int colBase = blockIdx.x * 64;

  v8f acc[2][2];
  for (int i = 0; i < 2; ++i)
    for (int j = 0; j < 2; ++j)
      for (int e = 0; e < 8; ++e) acc[i][j][e] = 0.f;

  for (int k0 = 0; k0 < K; k0 += 32) {
    {  // A tile: 128 rows x 32 k ; 16 elems (2 chunks) per thread
      int row = tid >> 1;
      int kk = (tid & 1) * 16;
      int gr = rowBase + row;
      v8bf c0, c1;
      if (A_BF16) {
        const bf16* src = (const bf16*)A_ + (size_t)gr * K + k0 + kk;
        if (gr < M) { c0 = *(const v8bf*)src; c1 = *(const v8bf*)(src + 8); }
        else        { c0 = zero8(); c1 = zero8(); }
      } else {
        const float* src = (const float*)A_ + (size_t)gr * K + k0 + kk;
        for (int j = 0; j < 8; ++j) {
          c0[j] = (gr < M) ? (bf16)src[j] : (bf16)0.f;
          c1[j] = (gr < M) ? (bf16)src[8 + j] : (bf16)0.f;
        }
      }
      *(v8bf*)&As[row][kk] = c0;
      *(v8bf*)&As[row][kk + 8] = c1;
    }
    {  // W tile transposed: Bs[n][k] = W[k][n] ; 8 elems per thread
      int n  = tid & 63;
      int kk = (tid >> 6) * 8;
      int gc = colBase + n;
      v8bf c;
      for (int j = 0; j < 8; ++j)
        c[j] = (gc < N) ? (bf16)W[(size_t)(k0 + kk + j) * N + gc] : (bf16)0.f;
      *(v8bf*)&Bs[n][kk] = c;
    }
    __syncthreads();

    v16bf af[2], bfr[2];
    for (int i = 0; i < 2; ++i) af[i]  = ld_frag(&As[wm * 32 + i * 16 + r][0], lh);
    for (int j = 0; j < 2; ++j) bfr[j] = ld_frag(&Bs[wn * 32 + j * 16 + r][0], lh);
    for (int i = 0; i < 2; ++i)
      for (int j = 0; j < 2; ++j)
        acc[i][j] = wmma_bf16(af[i], bfr[j], acc[i][j]);
    __syncthreads();
  }

  // epilogue; C layout: VGPR e <-> M = lh*8 + e, N = r (per 16x16 tile)
  for (int i = 0; i < 2; ++i) {
    for (int j = 0; j < 2; ++j) {
      for (int e = 0; e < 8; ++e) {
        int row = rowBase + wm * 32 + i * 16 + lh * 8 + e;
        int col = colBase + wn * 32 + j * 16 + r;
        float v = acc[i][j][e];
        if (col < N && bias) v += bias[col];
        if (ROPE) {
          int rr = (row < M) ? row : 0;
          int b = rr / NVOX, n = rr % NVOX;
          if (row < M && tgt_mask[b * NVOX + n] != 0) v += tgt_row[col];
          // rotate-half partner lives at col^1 -> adjacent lane
          float partner = __shfl_xor(v, 1, 32);
          int axis = col >> 8;
          float fr = inv_freq[(axis << 8) + (col & 255)];
          float p = (row < M) ? rel[((size_t)b * NVOX + n) * 3 + axis] : 0.f;
          float sn, cs;
          __sincosf(p * fr, &sn, &cs);
          float f1 = (col & 1) ? partner : -partner;
          float outv = v * cs + f1 * sn;
          if (row < M)
            ((bf16*)C_)[((size_t)(b * (2 * NVOX) + mem_off + n)) * HIDDIM + col] =
                (bf16)outv;
        } else if (row < M && col < N) {
          if (OUT_MODE == 2) v = fmaxf(v, 0.f);
          if (OUT_MODE == 1)
            ((bf16*)C_)[(size_t)row * N + col] = (bf16)v;
          else
            ((float*)C_)[(size_t)row * N + col] = v;
        }
      }
    }
  }
}

// ---------------------------------------------------------------------------
// Flash attention per (head, batch) block: Lq <= 128 queries resident in LDS,
// keys streamed in 32-wide chunks; S = QK^T and O += P@V via WMMA; online
// softmax in LDS. V chunk stored transposed so P@V B-frags are 2x b128 loads.
// ---------------------------------------------------------------------------
__global__ __launch_bounds__(256) void attn_kernel(
    const bf16* __restrict__ Qg, const bf16* __restrict__ Kg,
    const bf16* __restrict__ Vg, const unsigned char* __restrict__ kpad,
    float* __restrict__ Out, int Lq, int Lk) {
  const int hh = blockIdx.x;
  const int b  = blockIdx.y;
  const int tid = threadIdx.x;
  const int lane = tid & 31;
  const int wave = tid >> 5;
  const int r  = lane & 15;
  const int lh = lane >> 4;

  __shared__ __align__(16) bf16 Qs[128][72];   // [q][d], 144B rows
  __shared__ __align__(16) bf16 Ks[32][72];    // [key][d]
  __shared__ __align__(16) bf16 Vt[64][40];    // [d][key] (transposed)
  __shared__ __align__(16) float Ss[8][16][32];
  __shared__ float Ms[8][16], Ls[8][16], Cf[8][16];

  const bf16* Qp = Qg + (size_t)b * Lq * HIDDIM + hh * HDIM;
  const bf16* Kp = Kg + (size_t)b * Lk * HIDDIM + hh * HDIM;
  const bf16* Vp = Vg + (size_t)b * Lk * HIDDIM + hh * HDIM;

  // Q tile: 128x64, v8bf chunks (4 per thread)
  for (int id = tid; id < 128 * 8; id += 256) {
    int row = id >> 3;
    int c0 = (id & 7) * 8;
    v8bf q = (row < Lq) ? *(const v8bf*)(Qp + (size_t)row * HIDDIM + c0) : zero8();
    *(v8bf*)&Qs[row][c0] = q;
  }
  if (lane < 16) { Ms[wave][lane] = -1e30f; Ls[wave][lane] = 0.f; }

  v8f o[4];
  for (int t = 0; t < 4; ++t)
    for (int e = 0; e < 8; ++e) o[t][e] = 0.f;

  const int nch = (Lk + 31) >> 5;
  for (int ch = 0; ch < nch; ++ch) {
    __syncthreads();  // protect Ks/Vt/Ss reuse + initial Q load
    const int kb0 = ch * 32;
    {  // K row-major, V transposed; one v8bf chunk per thread
      int kr = tid >> 3;
      int c0 = (tid & 7) * 8;
      int kg = kb0 + kr;
      v8bf kv = zero8(), vv = zero8();
      if (kg < Lk) {
        kv = *(const v8bf*)(Kp + (size_t)kg * HIDDIM + c0);
        vv = *(const v8bf*)(Vp + (size_t)kg * HIDDIM + c0);
      }
      *(v8bf*)&Ks[kr][c0] = kv;
      for (int j = 0; j < 8; ++j) Vt[c0 + j][kr] = vv[j];
    }
    __syncthreads();

    // S(16x32) = Q_strip(16x64) @ K^T ; K-dim = head dim (2 steps of 32)
    v8f s[2];
    for (int t = 0; t < 2; ++t)
      for (int e = 0; e < 8; ++e) s[t][e] = 0.f;
    const int ar = wave * 16 + r;
    for (int kk = 0; kk < 64; kk += 32) {
      v16bf a = ld_frag(&Qs[ar][kk], lh);
      for (int t = 0; t < 2; ++t) {
        v16bf bb = ld_frag(&Ks[t * 16 + r][kk], lh);
        s[t] = wmma_bf16(a, bb, s[t]);
      }
    }
    // scale + mask -> LDS
    for (int t = 0; t < 2; ++t) {
      for (int e = 0; e < 8; ++e) {
        int m = lh * 8 + e;
        int kc = t * 16 + r;
        int kg = kb0 + kc;
        float v = s[t][e] * 0.125f;  // 1/sqrt(64)
        if (kg >= Lk || (kpad && kpad[(size_t)b * Lk + kg])) v = -1e9f;
        Ss[wave][m][kc] = v;
      }
    }
    __syncthreads();
    // online softmax (lanes 0..15 each own one query row of the strip)
    if (lh == 0) {
      float mo = Ms[wave][r], lo = Ls[wave][r];
      float mc = -1e30f;
      for (int kx = 0; kx < 32; ++kx) mc = fmaxf(mc, Ss[wave][r][kx]);
      float mn = fmaxf(mo, mc);
      float corr = __expf(mo - mn);
      float lsum = 0.f;
      for (int kx = 0; kx < 32; ++kx) {
        float p = __expf(Ss[wave][r][kx] - mn);
        Ss[wave][r][kx] = p;
        lsum += p;
      }
      Ms[wave][r] = mn;
      Ls[wave][r] = lo * corr + lsum;
      Cf[wave][r] = corr;
    }
    __syncthreads();
    // rescale accumulator, then O += P(16x32) @ V(32x64)
    for (int t = 0; t < 4; ++t)
      for (int e = 0; e < 8; ++e) o[t][e] *= Cf[wave][lh * 8 + e];

    v16bf a;
    {
      const float* p0 = &Ss[wave][r][lh * 8];
      const float* p1 = &Ss[wave][r][16 + lh * 8];
      for (int j = 0; j < 8; ++j) {
        a[j] = (bf16)p0[j];
        a[8 + j] = (bf16)p1[j];
      }
    }
    for (int t = 0; t < 4; ++t) {
      v16bf bb = ld_frag(&Vt[t * 16 + r][0], lh);
      o[t] = wmma_bf16(a, bb, o[t]);
    }
  }
  __syncthreads();
  for (int t = 0; t < 4; ++t) {
    for (int e = 0; e < 8; ++e) {
      int m = lh * 8 + e;
      int q = wave * 16 + m;
      if (q < Lq) {
        int c = t * 16 + r;
        float l = Ls[wave][m];
        Out[((size_t)b * Lq + q) * HIDDIM + hh * HDIM + c] =
            (l > 0.f) ? o[t][e] / l : 0.f;
      }
    }
  }
}

// ---------------------------------------------------------------------------
// Host orchestration (graph-capture safe: only async ops on `stream`)
// ---------------------------------------------------------------------------
extern "C" void kernel_launch(void* const* d_in, const int* in_sizes, int n_in,
                              void* d_out, int out_size, void* d_ws,
                              size_t ws_size, hipStream_t stream) {
  (void)in_sizes; (void)n_in; (void)out_size; (void)ws_size;
  const float* scene_voxs = (const float*)d_in[0];
  const int*   vox_masks  = (const int*)d_in[1];
  const float* vox3d      = (const float*)d_in[2];
  const float* vox2d      = (const float*)d_in[3];
  const int*   v2mask     = (const int*)d_in[4];
  const int*   tgt_mask   = (const int*)d_in[5];
  const float* pos        = (const float*)d_in[6];
  const float* ori        = (const float*)d_in[7];
  const float* txt_embed  = (const float*)d_in[8];
  const int*   txt_mask   = (const int*)d_in[9];
  const float* tgt_vox_w  = (const float*)d_in[10];
  const float* w3d  = (const float*)d_in[11];
  const float* b3d  = (const float*)d_in[12];
  const float* w2d  = (const float*)d_in[13];
  const float* b2d  = (const float*)d_in[14];
  const float* wtxt = (const float*)d_in[15];
  const float* btxt = (const float*)d_in[16];
  const float* qpos = (const float*)d_in[17];
  const float* invf = (const float*)d_in[18];
  const float* sa_w = (const float*)d_in[19];
  const float* sa_b = (const float*)d_in[20];
  const float* ct_w = (const float*)d_in[21];
  const float* ct_b = (const float*)d_in[22];
  const float* cv_w = (const float*)d_in[23];
  const float* cv_b = (const float*)d_in[24];
  const float* ffw1 = (const float*)d_in[25];
  const float* ffb1 = (const float*)d_in[26];
  const float* ffw2 = (const float*)d_in[27];
  const float* ffb2 = (const float*)d_in[28];
  const float* ln_s = (const float*)d_in[29];
  const float* ln_b = (const float*)d_in[30];

  char* base = (char*)d_ws;
  size_t off = 0;
  auto take = [&](size_t bytes) -> char* {
    char* p = base + off;
    off += (bytes + 255) & ~(size_t)255;
    return p;
  };
  const int Mq = BATCH * NQCNT;       // 640
  const int Mt = BATCH * TTXT;        // 512
  const int Mv = BATCH * NVOX;        // 16384
  const int Mm = BATCH * 2 * NVOX;    // 32768

  bf16*  memv  = (bf16*)take((size_t)Mm * HIDDIM * sizeof(bf16));  // 48 MB
  bf16*  Kbuf  = (bf16*)take((size_t)Mm * HIDDIM * sizeof(bf16));  // 48 MB
  bf16*  Vbuf  = (bf16*)take((size_t)Mm * HIDDIM * sizeof(bf16));  // 48 MB
  float* txt   = (float*)take((size_t)Mt * HIDDIM * sizeof(float));
  float* rel   = (float*)take((size_t)Mv * 3 * sizeof(float));
  float* Q     = (float*)take((size_t)Mq * HIDDIM * sizeof(float));
  float* qk    = (float*)take((size_t)Mq * HIDDIM * sizeof(float));
  bf16*  qb    = (bf16*)take((size_t)Mq * HIDDIM * sizeof(bf16));
  bf16*  kbS   = (bf16*)take((size_t)Mq * HIDDIM * sizeof(bf16));
  bf16*  vbS   = (bf16*)take((size_t)Mq * HIDDIM * sizeof(bf16));
  float* attno = (float*)take((size_t)Mq * HIDDIM * sizeof(float));
  float* tmp   = (float*)take((size_t)Mq * HIDDIM * sizeof(float));
  float* hbuf  = (float*)take((size_t)Mq * DFFDIM * sizeof(float));
  unsigned char* vox_kp = (unsigned char*)take((size_t)Mm);
  unsigned char* txt_kp = (unsigned char*)take((size_t)Mt);

  // ---- preprocessing ----
  rel_kernel<<<(Mv + 255) / 256, 256, 0, stream>>>(scene_voxs, pos, ori, rel, Mv);
  voxmask_kernel<<<(Mm + 255) / 256, 256, 0, stream>>>(vox_masks, v2mask, vox_kp, Mm);
  txtmask_kernel<<<(Mt + 255) / 256, 256, 0, stream>>>(txt_mask, txt_kp, Mt);

  const dim3 gv(HIDDIM / 64, (Mv + 127) / 128);
  gemm_kernel<0, 1, 1><<<gv, 256, 0, stream>>>(
      vox3d, w3d, b3d, memv, Mv, HIDDIM, HIDDIM, tgt_mask, tgt_vox_w, rel, invf, 0);
  gemm_kernel<0, 1, 1><<<gv, 256, 0, stream>>>(
      vox2d, w2d, b2d, memv, Mv, HIDDIM, HIDDIM, tgt_mask, tgt_vox_w, rel, invf, NVOX);

  const dim3 gt(HIDDIM / 64, (Mt + 127) / 128);
  gemm_kernel<0, 0, 0><<<gt, 256, 0, stream>>>(
      txt_embed, wtxt, btxt, txt, Mt, HIDDIM, HIDDIM,
      nullptr, nullptr, nullptr, nullptr, 0);

  hipMemsetAsync(Q, 0, (size_t)Mq * HIDDIM * sizeof(float), stream);

  const dim3 gq(HIDDIM / 64, (Mq + 127) / 128);    // (12,5)
  const dim3 gm(HIDDIM / 64, (Mm + 127) / 128);    // (12,256)
  const dim3 gf1(DFFDIM / 64, (Mq + 127) / 128);   // (48,5)
  const dim3 ga(NHEADS, BATCH);
  const int HH = HIDDIM * HIDDIM;

  for (int l = 0; l < NLAYERS; ++l) {
    const float* saW = sa_w + (size_t)l * 4 * HH;
    const float* saB = sa_b + (size_t)l * 4 * HIDDIM;
    const float* ctW = ct_w + (size_t)l * 4 * HH;
    const float* ctB = ct_b + (size_t)l * 4 * HIDDIM;
    const float* cvW = cv_w + (size_t)l * 4 * HH;
    const float* cvB = cv_b + (size_t)l * 4 * HIDDIM;
    const float* lnS = ln_s + (size_t)l * 4 * HIDDIM;
    const float* lnB = ln_b + (size_t)l * 4 * HIDDIM;

    // ---- self attention: q,k from (Q+qpos), v from Q ----
    addpos_kernel<<<(Mq * HIDDIM + 255) / 256, 256, 0, stream>>>(Q, qpos, qk, Mq * HIDDIM);
    gemm_kernel<0, 1, 0><<<gq, 256, 0, stream>>>(qk, saW + 0 * HH, saB + 0 * HIDDIM, qb,
        Mq, HIDDIM, HIDDIM, nullptr, nullptr, nullptr, nullptr, 0);
    gemm_kernel<0, 1, 0><<<gq, 256, 0, stream>>>(qk, saW + 1 * HH, saB + 1 * HIDDIM, kbS,
        Mq, HIDDIM, HIDDIM, nullptr, nullptr, nullptr, nullptr, 0);
    gemm_kernel<0, 1, 0><<<gq, 256, 0, stream>>>(Q, saW + 2 * HH, saB + 2 * HIDDIM, vbS,
        Mq, HIDDIM, HIDDIM, nullptr, nullptr, nullptr, nullptr, 0);
    attn_kernel<<<ga, 256, 0, stream>>>(qb, kbS, vbS, nullptr, attno, NQCNT, NQCNT);
    gemm_kernel<0, 0, 0><<<gq, 256, 0, stream>>>(attno, saW + 3 * HH, saB + 3 * HIDDIM, tmp,
        Mq, HIDDIM, HIDDIM, nullptr, nullptr, nullptr, nullptr, 0);
    add_ln_kernel<<<Mq, 256, 0, stream>>>(Q, tmp, lnS + 0 * HIDDIM, lnB + 0 * HIDDIM);

    // ---- cross attention to text ----
    addpos_kernel<<<(Mq * HIDDIM + 255) / 256, 256, 0, stream>>>(Q, qpos, qk, Mq * HIDDIM);
    gemm_kernel<0, 1, 0><<<gq, 256, 0, stream>>>(qk, ctW + 0 * HH, ctB + 0 * HIDDIM, qb,
        Mq, HIDDIM, HIDDIM, nullptr, nullptr, nullptr, nullptr, 0);
    gemm_kernel<0, 1, 0><<<gt, 256, 0, stream>>>(txt, ctW + 1 * HH, ctB + 1 * HIDDIM, kbS,
        Mt, HIDDIM, HIDDIM, nullptr, nullptr, nullptr, nullptr, 0);
    gemm_kernel<0, 1, 0><<<gt, 256, 0, stream>>>(txt, ctW + 2 * HH, ctB + 2 * HIDDIM, vbS,
        Mt, HIDDIM, HIDDIM, nullptr, nullptr, nullptr, nullptr, 0);
    attn_kernel<<<ga, 256, 0, stream>>>(qb, kbS, vbS, txt_kp, attno, NQCNT, TTXT);
    gemm_kernel<0, 0, 0><<<gq, 256, 0, stream>>>(attno, ctW + 3 * HH, ctB + 3 * HIDDIM, tmp,
        Mq, HIDDIM, HIDDIM, nullptr, nullptr, nullptr, nullptr, 0);
    add_ln_kernel<<<Mq, 256, 0, stream>>>(Q, tmp, lnS + 1 * HIDDIM, lnB + 1 * HIDDIM);

    // ---- cross attention to memory_vox (dominant compute) ----
    addpos_kernel<<<(Mq * HIDDIM + 255) / 256, 256, 0, stream>>>(Q, qpos, qk, Mq * HIDDIM);
    gemm_kernel<0, 1, 0><<<gq, 256, 0, stream>>>(qk, cvW + 0 * HH, cvB + 0 * HIDDIM, qb,
        Mq, HIDDIM, HIDDIM, nullptr, nullptr, nullptr, nullptr, 0);
    gemm_kernel<1, 1, 0><<<gm, 256, 0, stream>>>(memv, cvW + 1 * HH, cvB + 1 * HIDDIM, Kbuf,
        Mm, HIDDIM, HIDDIM, nullptr, nullptr, nullptr, nullptr, 0);
    gemm_kernel<1, 1, 0><<<gm, 256, 0, stream>>>(memv, cvW + 2 * HH, cvB + 2 * HIDDIM, Vbuf,
        Mm, HIDDIM, HIDDIM, nullptr, nullptr, nullptr, nullptr, 0);
    attn_kernel<<<ga, 256, 0, stream>>>(qb, Kbuf, Vbuf, vox_kp, attno, NQCNT, 2 * NVOX);
    gemm_kernel<0, 0, 0><<<gq, 256, 0, stream>>>(attno, cvW + 3 * HH, cvB + 3 * HIDDIM, tmp,
        Mq, HIDDIM, HIDDIM, nullptr, nullptr, nullptr, nullptr, 0);
    add_ln_kernel<<<Mq, 256, 0, stream>>>(Q, tmp, lnS + 2 * HIDDIM, lnB + 2 * HIDDIM);

    // ---- FFN ----
    gemm_kernel<0, 2, 0><<<gf1, 256, 0, stream>>>(Q, ffw1 + (size_t)l * HIDDIM * DFFDIM,
        ffb1 + (size_t)l * DFFDIM, hbuf, Mq, DFFDIM, HIDDIM,
        nullptr, nullptr, nullptr, nullptr, 0);
    gemm_kernel<0, 0, 0><<<gq, 256, 0, stream>>>(hbuf, ffw2 + (size_t)l * DFFDIM * HIDDIM,
        ffb2 + (size_t)l * HIDDIM, tmp, Mq, HIDDIM, DFFDIM,
        nullptr, nullptr, nullptr, nullptr, 0);
    add_ln_kernel<<<Mq, 256, 0, stream>>>(Q, tmp, lnS + 3 * HIDDIM, lnB + 3 * HIDDIM);
  }

  hipMemcpyAsync(d_out, Q, (size_t)Mq * HIDDIM * sizeof(float),
                 hipMemcpyDeviceToDevice, stream);
}
